// FastPointTransformer_44607530336588
// MI455X (gfx1250) — matrix operand order repaired and checked by reading, not compiled, not run on hardware
//
#include <hip/hip_runtime.h>
#include <hip/hip_bf16.h>
#include <hip/hip_fp16.h>

typedef __attribute__((ext_vector_type(16))) _Float16 v16h;
typedef __attribute__((ext_vector_type(8)))  _Float16 v8h;
typedef __attribute__((ext_vector_type(4)))  _Float16 v4h;
typedef __attribute__((ext_vector_type(8)))  float    v8f;

#define NPTS 131072
#define CCH  128
#define HH   8
#define DD   16
#define KK   27
#define MM   2097152

// ---------------------------------------------------------------------------
// Weight transpose + f16 convert: Bt[mat][n][k] = W[mat][k][n]
// ---------------------------------------------------------------------------
__global__ void prep_weights(const float* __restrict__ Wq, const float* __restrict__ Wv,
                             const float* __restrict__ W3, const float* __restrict__ Wo,
                             _Float16* __restrict__ wt) {
    int id  = blockIdx.x * blockDim.x + threadIdx.x;   // 4*128*128
    int mat = id >> 14;
    int rem = id & 16383;
    int n   = rem >> 7;
    int k   = rem & 127;
    const float* src = (mat == 0) ? Wq : (mat == 1) ? Wv : (mat == 2) ? W3 : Wo;
    wt[mat * 16384 + n * 128 + k] = (_Float16)src[k * 128 + n];
}

// ---------------------------------------------------------------------------
// npe = l2norm(inter_pos_enc) over D, stored f16. 27*8 rows of 16.
// ---------------------------------------------------------------------------
__global__ void prep_npe(const float* __restrict__ pe, _Float16* __restrict__ npe) {
    int id = threadIdx.x;
    if (id < KK * HH) {
        const float* s = pe + id * DD;
        float t[DD];
        float ss = 0.f;
        #pragma unroll
        for (int d = 0; d < DD; ++d) { t[d] = s[d]; ss += t[d] * t[d]; }
        float inv = 1.f / fmaxf(sqrtf(ss), 1e-12f);
        #pragma unroll
        for (int d = 0; d < DD; ++d) npe[id * DD + d] = (_Float16)(t[d] * inv);
    }
}

// ---------------------------------------------------------------------------
// Per-point MLP with inference BatchNorm: h2 = relu(bn2(relu(bn1(p@W1))@W2))
// One thread per (n, c).
// ---------------------------------------------------------------------------
__global__ void mlp_kernel(const float* __restrict__ pts, const float* __restrict__ W1,
                           const float* __restrict__ g1, const float* __restrict__ b1,
                           const float* __restrict__ m1, const float* __restrict__ v1,
                           const float* __restrict__ W2,
                           const float* __restrict__ g2, const float* __restrict__ b2,
                           const float* __restrict__ m2, const float* __restrict__ v2,
                           _Float16* __restrict__ h2out) {
    int id = blockIdx.x * blockDim.x + threadIdx.x;    // n*128 + c
    int n  = id >> 7;
    int c  = id & 127;
    float p0 = pts[n * 3 + 0], p1 = pts[n * 3 + 1], p2 = pts[n * 3 + 2];
    float h1[3];
    #pragma unroll
    for (int j = 0; j < 3; ++j) {
        float a = p0 * W1[0 * 3 + j] + p1 * W1[1 * 3 + j] + p2 * W1[2 * 3 + j];
        a = (a - m1[j]) * g1[j] * rsqrtf(v1[j] + 1e-5f) + b1[j];
        h1[j] = fmaxf(a, 0.f);
    }
    float a = h1[0] * W2[0 * CCH + c] + h1[1] * W2[1 * CCH + c] + h1[2] * W2[2 * CCH + c];
    a = (a - m2[c]) * g2[c] * rsqrtf(v2[c] + 1e-5f) + b2[c];
    h2out[id] = (_Float16)fmaxf(a, 0.f);
}

// ---------------------------------------------------------------------------
// A-tile loader (wave32, 16x16x32 f16)
// A 16x32: lanes 0-15 -> M=lane, K 0..15 (elems 0-15);
//          lanes 16-31 -> same M, K 16..31.
// ---------------------------------------------------------------------------
__device__ inline v16h load_atile(const _Float16* __restrict__ base, int row0, int k0, int lane) {
    int r = row0 + (lane & 15);
    int k = k0 + ((lane >> 4) << 4);
    const _Float16* p = base + (size_t)r * CCH + k;
    v8h lo = *(const v8h*)p;
    v8h hi = *(const v8h*)(p + 8);
    v16h t;
    #pragma unroll
    for (int i = 0; i < 8; ++i) { t[i] = lo[i]; t[8 + i] = hi[i]; }
    return t;
}

// ---------------------------------------------------------------------------
// GEMM: out[N,128] = A[N,128](f16) @ Bt^T + bias, epilogue per MODE.
// Weights staged once per block into LDS (32 KB of 320 KB/WGP), then B-tiles
// come from ds_load_b128: lane -> column n0+(lane&15); lanes 0-15 hold
// K=k0..k0+15, lanes 16-31 hold K=k0+16..k0+31 (contiguous 32 B per lane).
// MODE 0: += resid (f32), store f16     (x = feats + h2@W3 + b3)
// MODE 1: per-head l2norm, store f16    (nq)
// MODE 2: plain, store f16              (v)
// MODE 3: plain, store f32 to d_out     (final @ Wo)
// Block = 256 threads = 8 waves; wave owns 16 rows x 128 cols.
// ---------------------------------------------------------------------------
template <int MODE>
__global__ void gemm_nc(const _Float16* __restrict__ A, const _Float16* __restrict__ Bt,
                        const float* __restrict__ bias, const float* __restrict__ resid,
                        _Float16* __restrict__ outH, float* __restrict__ outF) {
    __shared__ _Float16 sB[CCH * CCH];                 // 32 KB
    {
        const v8h* src = (const v8h*)Bt;
        v8h* dst = (v8h*)sB;
        #pragma unroll
        for (int i = 0; i < 8; ++i)
            dst[threadIdx.x + 256 * i] = src[threadIdx.x + 256 * i];
    }
    __syncthreads();

    int lane = threadIdx.x & 31;
    int wave = threadIdx.x >> 5;
    int row0 = blockIdx.x * 128 + wave * 16;

    v8f acc[8];
    #pragma unroll
    for (int i = 0; i < 8; ++i)
        #pragma unroll
        for (int j = 0; j < 8; ++j) acc[i][j] = 0.f;

    int bcol = lane & 15;
    int bkofs = (lane >> 4) << 4;
    #pragma unroll
    for (int kt = 0; kt < 4; ++kt) {
        v16h a = load_atile(A, row0, kt * 32, lane);
        #pragma unroll
        for (int nt = 0; nt < 8; ++nt) {
            v16h b = *(const v16h*)(sB + (size_t)(nt * 16 + bcol) * CCH + kt * 32 + bkofs);
            acc[nt] = __builtin_amdgcn_wmma_f32_16x16x32_f16(
                false, a, false, b, (short)0, acc[nt], false, false);
        }
    }

    // C/D layout: VGPR r -> row row0 + r + (lane>=16 ? 8 : 0), col nt*16 + (lane&15)
    int colL   = lane & 15;
    int rowAdd = (lane >> 4) << 3;
    #pragma unroll
    for (int nt = 0; nt < 8; ++nt) {
        #pragma unroll
        for (int r = 0; r < 8; ++r) {
            int row = row0 + rowAdd + r;
            int col = nt * 16 + colL;
            size_t idx = (size_t)row * CCH + col;
            float val = acc[nt][r] + bias[col];
            if (MODE == 0) {
                val += resid[idx];
                outH[idx] = (_Float16)val;
            } else if (MODE == 1) {
                // per-head l2norm: 16 cols of this row live in this 16-lane half
                float ss = val * val;
                ss += __shfl_xor(ss, 1, 32);
                ss += __shfl_xor(ss, 2, 32);
                ss += __shfl_xor(ss, 4, 32);
                ss += __shfl_xor(ss, 8, 32);
                float nrm = fmaxf(sqrtf(ss), 1e-12f);
                outH[idx] = (_Float16)(val / nrm);
            } else if (MODE == 2) {
                outH[idx] = (_Float16)val;
            } else {
                outF[idx] = val;
            }
        }
    }
}

// ---------------------------------------------------------------------------
// Pair attention: attn[m,h] = <nq[qi,h,:], npe[kn,h,:]>; out[qi] += attn*v[ki]
// One wave per pair; lane handles 4 contiguous halves (e0 = lane*4), so a
// head's 16 d-values live in 4 consecutive lanes -> shfl_xor masks 1,2.
// Gathers are single b64 loads; 32 lanes coalesce a full 256 B row.
// ---------------------------------------------------------------------------
__global__ void pair_kernel(const _Float16* __restrict__ nq, const _Float16* __restrict__ npe,
                            const _Float16* __restrict__ vmat, const int* __restrict__ qidx,
                            const int* __restrict__ kidx, const int* __restrict__ kern,
                            float* __restrict__ out) {
    int lane = threadIdx.x & 31;
    int wave = threadIdx.x >> 5;
    int m = blockIdx.x * 8 + wave;
    int qi = qidx[m];
    int ki = kidx[m];
    int kn = kern[m];
    int e0 = lane * 4;

    v4h q4 = *(const v4h*)(nq  + (size_t)qi * CCH + e0);
    v4h p4 = *(const v4h*)(npe + (size_t)kn * CCH + e0);
    float prod = (float)q4[0] * (float)p4[0] + (float)q4[1] * (float)p4[1]
               + (float)q4[2] * (float)p4[2] + (float)q4[3] * (float)p4[3];
    // reduce over the 4 lanes sharing this head
    prod += __shfl_xor(prod, 1, 32);
    prod += __shfl_xor(prod, 2, 32);

    v4h v4 = *(const v4h*)(vmat + (size_t)ki * CCH + e0);
    float* o = out + (size_t)qi * CCH + e0;
    atomicAdd(o + 0, prod * (float)v4[0]);
    atomicAdd(o + 1, prod * (float)v4[1]);
    atomicAdd(o + 2, prod * (float)v4[2]);
    atomicAdd(o + 3, prod * (float)v4[3]);
}

// ---------------------------------------------------------------------------
__global__ void zero_f32(float* __restrict__ p) {
    int id = blockIdx.x * blockDim.x + threadIdx.x;   // one float4 per thread
    ((float4*)p)[id] = make_float4(0.f, 0.f, 0.f, 0.f);
}

__global__ void f32_to_f16(const float* __restrict__ s, _Float16* __restrict__ d) {
    int id = (blockIdx.x * blockDim.x + threadIdx.x) * 4;
    #pragma unroll
    for (int i = 0; i < 4; ++i) d[id + i] = (_Float16)s[id + i];
}

// ---------------------------------------------------------------------------
extern "C" void kernel_launch(void* const* d_in, const int* in_sizes, int n_in,
                              void* d_out, int out_size, void* d_ws, size_t ws_size,
                              hipStream_t stream) {
    const float* feats = (const float*)d_in[0];
    const float* pts   = (const float*)d_in[1];
    const float* Wq    = (const float*)d_in[2];
    const float* bq    = (const float*)d_in[3];
    const float* Wv    = (const float*)d_in[4];
    const float* bv    = (const float*)d_in[5];
    const float* Wo    = (const float*)d_in[6];
    const float* bo    = (const float*)d_in[7];
    const float* pe    = (const float*)d_in[8];
    const float* W1    = (const float*)d_in[9];
    const float* g1    = (const float*)d_in[10];
    const float* b1    = (const float*)d_in[11];
    const float* m1    = (const float*)d_in[12];
    const float* v1    = (const float*)d_in[13];
    const float* W2    = (const float*)d_in[14];
    const float* g2    = (const float*)d_in[15];
    const float* b2    = (const float*)d_in[16];
    const float* m2    = (const float*)d_in[17];
    const float* v2    = (const float*)d_in[18];
    const float* W3    = (const float*)d_in[19];
    const float* b3    = (const float*)d_in[20];
    const int* qidx    = (const int*)d_in[21];
    const int* kidx    = (const int*)d_in[22];
    const int* kern    = (const int*)d_in[23];
    float* out         = (float*)d_out;

    // Workspace layout
    const size_t HBUF = (size_t)NPTS * CCH * 2;            // 32 MB per f16 buffer
    char* ws = (char*)d_ws;
    _Float16* h2f  = (_Float16*)(ws);                      // h2, reused as out_f16
    _Float16* xf   = (_Float16*)(ws + HBUF);               // x f16
    _Float16* nqf  = (_Float16*)(ws + 2 * HBUF);           // normalized q f16
    _Float16* vf   = (_Float16*)(ws + 3 * HBUF);           // v f16
    float*    accf = (float*)   (ws + 4 * HBUF);           // out accumulator f32 (64 MB)
    _Float16* wtf  = (_Float16*)(ws + 4 * HBUF + (size_t)NPTS * CCH * 4);
    _Float16* npef = wtf + 4 * CCH * CCH;                  // 27*128 f16

    _Float16* WtQ = wtf;
    _Float16* WtV = wtf + 1 * CCH * CCH;
    _Float16* Wt3 = wtf + 2 * CCH * CCH;
    _Float16* WtO = wtf + 3 * CCH * CCH;

    // 1. weights -> transposed f16; pos-enc -> normalized f16
    prep_weights<<<(4 * CCH * CCH) / 256, 256, 0, stream>>>(Wq, Wv, W3, Wo, wtf);
    prep_npe<<<1, 256, 0, stream>>>(pe, npef);

    // 2. per-point MLP -> h2 f16
    mlp_kernel<<<(NPTS * CCH) / 256, 256, 0, stream>>>(pts, W1, g1, b1, m1, v1,
                                                       W2, g2, b2, m2, v2, h2f);

    // 3. x = feats + h2 @ W3 + b3          (WMMA, residual)
    gemm_nc<0><<<NPTS / 128, 256, 0, stream>>>(h2f, Wt3, b3, feats, xf, nullptr);
    // 4. nq = l2norm_perhead(x @ Wq + bq)  (WMMA, norm epilogue)
    gemm_nc<1><<<NPTS / 128, 256, 0, stream>>>(xf, WtQ, bq, nullptr, nqf, nullptr);
    // 5. v = x @ Wv + bv                   (WMMA)
    gemm_nc<2><<<NPTS / 128, 256, 0, stream>>>(xf, WtV, bv, nullptr, vf, nullptr);

    // 6. zero accumulator, gather/dot/scatter over M pairs (one wave per pair)
    zero_f32<<<(NPTS * CCH / 4) / 256, 256, 0, stream>>>(accf);
    pair_kernel<<<MM / 8, 256, 0, stream>>>(nqf, npef, vf, qidx, kidx, kern, accf);

    // 7. final: out = acc @ Wo + bo        (WMMA, f32 store)
    f32_to_f16<<<(NPTS * CCH / 4) / 256, 256, 0, stream>>>(accf, h2f);
    gemm_nc<3><<<NPTS / 128, 256, 0, stream>>>(h2f, WtO, bo, nullptr, nullptr, out);
}
